// TransformerXLBlock_11098195492878
// MI455X (gfx1250) — matrix operand order repaired and checked by reading, not compiled, not run on hardware
//
#include <hip/hip_runtime.h>

// ---------------------------------------------------------------------------
// Transformer-XL block for MI455X (gfx1250, wave32, WMMA + async-to-LDS + TDM).
// All GEMMs in bf16 with f32 accumulate via v_wmma_f32_16x16x32_bf16.
// ---------------------------------------------------------------------------

typedef __attribute__((ext_vector_type(16))) __bf16 v16bf;
typedef __attribute__((ext_vector_type(8)))  float  v8f;

union FragU { v16bf v; unsigned u[8]; };

static constexpr int kQ    = 1024;
static constexpr int kB    = 4;
static constexpr int kDM   = 1024;
static constexpr int kH    = 16;
static constexpr int kHD   = 64;
static constexpr int kMEM  = 512;
static constexpr int kKLEN = 1536;   // kMEM + kQ
static constexpr int kMLP  = 4096;
static constexpr float kSCALE = 0.125f;   // 1/sqrt(64)

// ---------------- CDNA5 async / TDM feature detection ----------------------
#if __has_builtin(__builtin_amdgcn_global_load_async_to_lds_b128)
#define HAVE_ASYNC 1
#pragma message("CDNA5: global_load_async_to_lds_b128 ENABLED")
#else
#define HAVE_ASYNC 0
#pragma message("CDNA5: async-to-LDS builtin NOT available, sync fallback")
#endif

#if __has_builtin(__builtin_amdgcn_tensor_load_to_lds)
#define HAVE_TDM 1
#pragma message("CDNA5: tensor_load_to_lds (TDM) ENABLED")
#else
#define HAVE_TDM 0
#pragma message("CDNA5: TDM builtin NOT available, per-thread staging fallback")
#endif

// Parameter type of the async builtins per hipcc diagnostic:
// 'int __attribute__((vector_size(16))) *' (generic address space).
typedef int v4i_vs __attribute__((vector_size(16)));

__device__ __forceinline__ void cp_async_b128(void* lds, const void* g) {
#if HAVE_ASYNC
  __builtin_amdgcn_global_load_async_to_lds_b128((v4i_vs*)g, (v4i_vs*)lds, 0, 0);
#else
  *(uint4*)lds = *(const uint4*)g;
#endif
}

__device__ __forceinline__ void async_wait() {
#if HAVE_ASYNC
#if __has_builtin(__builtin_amdgcn_s_wait_asynccnt)
  __builtin_amdgcn_s_wait_asynccnt(0);
#else
  asm volatile("s_wait_asynccnt 0" ::: "memory");
#endif
#endif
}

#if HAVE_TDM
typedef __attribute__((ext_vector_type(4))) unsigned int u32x4;
typedef __attribute__((ext_vector_type(4))) int i32x4;
typedef __attribute__((ext_vector_type(8))) int i32x8;

// 2D tile load (bf16 elements) via the Tensor Data Mover.
// Descriptor layout per CDNA5 ISA sec 8.3/8.4 (D# group0/group1).
__device__ __forceinline__ void tdm_load_tile_bf16(
    const void* g, void* lds, unsigned tile_w, unsigned tile_h,
    unsigned tensor_w, unsigned tensor_h, unsigned stride_elems) {
  unsigned long long ga = (unsigned long long)g;
  unsigned lo = (unsigned)(unsigned long long)lds;  // LDS byte offset (low bits)
  u32x4 g0;
  g0[0] = 1u;                                   // count=1, user mode
  g0[1] = lo;                                   // lds_addr (bytes)
  g0[2] = (unsigned)ga;                         // global_addr[31:0]
  g0[3] = (unsigned)((ga >> 32) & 0x01FFFFFFull) | (2u << 30);  // type=2
  i32x8 g1;
  g1[0] = (int)(1u << 16);                      // data_size=1 (2 bytes)
  g1[1] = (int)((tensor_w & 0xFFFFu) << 16);    // tensor_dim0[15:0]
  g1[2] = (int)(((tensor_w >> 16) & 0xFFFFu) | ((tensor_h & 0xFFFFu) << 16));
  g1[3] = (int)(((tensor_h >> 16) & 0xFFFFu) | ((tile_w & 0xFFFFu) << 16));
  g1[4] = (int)(tile_h & 0xFFFFu);              // tile_dim1 (tile_dim2=0)
  g1[5] = (int)stride_elems;                    // tensor_dim0_stride[31:0]
  g1[6] = 0;
  g1[7] = 0;
  i32x4 z4 = {0, 0, 0, 0};
#if defined(__clang_major__) && __clang_major__ >= 23
  i32x8 z8 = {0, 0, 0, 0, 0, 0, 0, 0};
  __builtin_amdgcn_tensor_load_to_lds(g0, g1, z4, z4, z8, 0);
#else
  __builtin_amdgcn_tensor_load_to_lds(g0, g1, z4, z4, 0);
#endif
}
#endif  // HAVE_TDM

__device__ __forceinline__ void tensor_wait() {
#if HAVE_TDM
  if ((threadIdx.x >> 5) == 0) __builtin_amdgcn_s_wait_tensorcnt(0);
#endif
}

// ---------------- WMMA helpers ---------------------------------------------
__device__ __forceinline__ v8f wmma_bf16(v16bf a, v16bf b, v8f c) {
  return __builtin_amdgcn_wmma_f32_16x16x32_bf16(false, a, false, b,
                                                 (short)0, c, false, false);
}

// A-matrix fragment (16x32 bf16), ISA 7.12.2 layout.
__device__ __forceinline__ v16bf frag_a(const __bf16* p, int row0, int stride, int col) {
  const int lane = threadIdx.x & 31;
  const int lr = lane & 15, hi = lane >> 4;
  FragU f;
  const __bf16* base = p + (row0 + lr) * stride + col + hi * 8;
#pragma unroll
  for (int d = 0; d < 8; ++d) {
    const int kb = ((d & 4) << 2) + ((d & 3) << 1);  // {0,2,4,6,16,18,20,22}
    f.u[d] = *(const unsigned*)(base + kb);
  }
  return f.v;
}

// B-matrix fragment (32x16 bf16) from LDS stored as [N][stride] (B^T).
__device__ __forceinline__ v16bf frag_b(const __bf16* p, int n0, int stride, int col) {
  const int lane = threadIdx.x & 31;
  const int lr = lane & 15, hi = lane >> 4;
  FragU f;
  const __bf16* base = p + (n0 + lr) * stride + col + hi * 8;
#pragma unroll
  for (int d = 0; d < 8; ++d)
    f.u[d] = *(const unsigned*)(base + ((d & 4) << 2) + ((d & 3) << 1));
  return f.v;
}

__device__ __forceinline__ float redmax16(float x) {
#pragma unroll
  for (int m = 1; m < 16; m <<= 1) x = fmaxf(x, __shfl_xor(x, m, 32));
  return x;
}
__device__ __forceinline__ float redsum16(float x) {
#pragma unroll
  for (int m = 1; m < 16; m <<= 1) x += __shfl_xor(x, m, 32);
  return x;
}

// ---------------------------------------------------------------------------
// f32 -> bf16 convert (flat)
// ---------------------------------------------------------------------------
__global__ __launch_bounds__(256) void cvt_bf16_k(const float* __restrict__ in,
                                                  __bf16* __restrict__ out, int n) {
  int i = blockIdx.x * 256 + threadIdx.x;
  if (i < n) out[i] = (__bf16)in[i];
}

// ---------------------------------------------------------------------------
// f32 [K,N] -> bf16 transposed [N,K] (32x32 LDS tile transpose)
// ---------------------------------------------------------------------------
__global__ __launch_bounds__(256) void transpose_cvt_k(const float* __restrict__ in,
                                                       __bf16* __restrict__ out,
                                                       int K, int N) {
  __shared__ __bf16 tile[32][33];
  const int kb = blockIdx.y * 32, nb = blockIdx.x * 32;
  const int tx = threadIdx.x & 31, ty = threadIdx.x >> 5;
#pragma unroll
  for (int r = ty; r < 32; r += 8)
    tile[r][tx] = (__bf16)in[(size_t)(kb + r) * N + nb + tx];
  __syncthreads();
#pragma unroll
  for (int r = ty; r < 32; r += 8)
    out[(size_t)(nb + r) * K + kb + tx] = tile[tx][r];
}

// ---------------------------------------------------------------------------
// Tiled bf16 GEMM: C[M,N] = A[M,K] @ Bt[N,K]^T (+bias) (+relu)
// 128x128 tile, K-step 32, double-buffered LDS, async/TDM staging.
// 8 waves (4x2), each wave 32x64 = 8 WMMA accumulators.
// ---------------------------------------------------------------------------
__device__ __forceinline__ void gemm_stage(const __bf16* __restrict__ A,
                                           const __bf16* __restrict__ Bt,
                                           __bf16* sAb, __bf16* sBb,
                                           int m0, int n0, int K, int k0) {
  const int t = threadIdx.x;
  const int row = t >> 1, half = t & 1;
#if HAVE_TDM
  if ((t >> 5) == 0)   // wave 0 issues one DMA for the whole 128x32 A tile
    tdm_load_tile_bf16(A + (size_t)m0 * K + k0, sAb, 32, 128, (unsigned)K, 128,
                       (unsigned)K);
#else
  {
    const __bf16* g = A + (size_t)(m0 + row) * K + k0 + half * 16;
    __bf16* l = sAb + row * 32 + half * 16;
    cp_async_b128(l, g);
    cp_async_b128(l + 8, g + 8);
  }
#endif
  {
    const __bf16* g = Bt + (size_t)(n0 + row) * K + k0 + half * 16;
    __bf16* l = sBb + row * 32 + half * 16;
    cp_async_b128(l, g);
    cp_async_b128(l + 8, g + 8);
  }
}

template <int OUTBF, int HASBIAS, int RELU>
__global__ __launch_bounds__(256) void gemm_bf16_k(
    const __bf16* __restrict__ A, const __bf16* __restrict__ Bt,
    const float* __restrict__ bias, float* __restrict__ Cf,
    __bf16* __restrict__ Cb, int M, int N, int K) {
  __shared__ __bf16 sA[2][128][32];
  __shared__ __bf16 sB[2][128][32];   // [n][k]

  const int m0 = blockIdx.y * 128;
  const int n0 = blockIdx.x * 128;
  const int t  = threadIdx.x;
  const int wid = t >> 5;
  const int wm = wid & 3, wn = wid >> 2;
  const int lane = t & 31, lr = lane & 15, hi = lane >> 4;

  v8f acc[2][4];
#pragma unroll
  for (int i = 0; i < 2; ++i)
#pragma unroll
    for (int j = 0; j < 4; ++j)
#pragma unroll
      for (int e = 0; e < 8; ++e) acc[i][j][e] = 0.f;

  gemm_stage(A, Bt, &sA[0][0][0], &sB[0][0][0], m0, n0, K, 0);

  for (int k0 = 0; k0 < K; k0 += 32) {
    const int cur = (k0 >> 5) & 1;
    tensor_wait();
    async_wait();
    __syncthreads();
    if (k0 + 32 < K)
      gemm_stage(A, Bt, &sA[cur ^ 1][0][0], &sB[cur ^ 1][0][0], m0, n0, K,
                 k0 + 32);

    v16bf af0 = frag_a(&sA[cur][0][0], wm * 32,      32, 0);
    v16bf af1 = frag_a(&sA[cur][0][0], wm * 32 + 16, 32, 0);
#pragma unroll
    for (int j = 0; j < 4; ++j) {
      v16bf bf = frag_b(&sB[cur][0][0], wn * 64 + j * 16, 32, 0);
      acc[0][j] = wmma_bf16(af0, bf, acc[0][j]);
      acc[1][j] = wmma_bf16(af1, bf, acc[1][j]);
    }
  }

#pragma unroll
  for (int i2 = 0; i2 < 2; ++i2)
#pragma unroll
    for (int j = 0; j < 4; ++j)
#pragma unroll
      for (int r = 0; r < 8; ++r) {
        const int m = m0 + wm * 32 + i2 * 16 + r + 8 * hi;
        const int n = n0 + wn * 64 + j * 16 + lr;
        float v = acc[i2][j][r];
        if (HASBIAS) v += bias[n];
        if (RELU) v = fmaxf(v, 0.f);
        if (OUTBF)
          Cb[(size_t)m * N + n] = (__bf16)v;
        else
          Cf[(size_t)m * N + n] = v;
      }
}

// ---------------------------------------------------------------------------
// Flash attention with Transformer-XL relative shift.
// Grid: (Q/64, B*H). Block: 128 threads = 4 waves, each wave owns 16 q-rows.
// BD[i,j] = (q_i+v) . rproj[j + Q-1-i]  (per-tile 128-row diagonal band)
// ---------------------------------------------------------------------------
__global__ __launch_bounds__(128) void attn_k(
    const __bf16* __restrict__ qkv,    // [KLEN*B, 3*H*HD] bf16
    const __bf16* __restrict__ rproj,  // [KLEN, H*HD] bf16
    const float* __restrict__ u, const float* __restrict__ vv,
    __bf16* __restrict__ ctx) {        // [Q*B, H*HD] bf16
  __shared__ __bf16 sK[64][64];        // key tile  [key][dim]  (B-layout)
  __shared__ __bf16 sVt[64][64];       // value tile [dim][key] (B-layout)
  __shared__ __bf16 sR[128][64];       // r band    [band][dim] (B-layout)
  __shared__ float  sT[4][16][128];    // per-wave BD tile bounce
  __shared__ __bf16 sP[4][16][64];     // per-wave P bounce

  const int i0 = blockIdx.x * 64;
  const int bh = blockIdx.y;
  const int b = bh >> 4;
  const int h = bh & 15;
  const int t = threadIdx.x;
  const int w = t >> 5, lane = t & 31, lr = lane & 15, hi = lane >> 4;

  // ---- stage q+u then q+v through sR, capture A-frags (K=64 -> 2 chunks) ----
  v16bf quf[2], qvf[2];
  {
    const int ii = t >> 1, half = t & 1;
    const size_t base =
        ((size_t)(kMEM + i0 + ii) * kB + b) * 3072 + h * 64 + half * 32;
#pragma unroll
    for (int dd = 0; dd < 32; ++dd) {
      float q = (float)qkv[base + dd] + u[h * 64 + half * 32 + dd];
      sR[ii][half * 32 + dd] = (__bf16)q;
    }
  }
  __syncthreads();
  quf[0] = frag_a(&sR[0][0], w * 16, 64, 0);
  quf[1] = frag_a(&sR[0][0], w * 16, 64, 32);
  __syncthreads();
  {
    const int ii = t >> 1, half = t & 1;
    const size_t base =
        ((size_t)(kMEM + i0 + ii) * kB + b) * 3072 + h * 64 + half * 32;
#pragma unroll
    for (int dd = 0; dd < 32; ++dd) {
      float q = (float)qkv[base + dd] + vv[h * 64 + half * 32 + dd];
      sR[ii][half * 32 + dd] = (__bf16)q;
    }
  }
  __syncthreads();
  qvf[0] = frag_a(&sR[0][0], w * 16, 64, 0);
  qvf[1] = frag_a(&sR[0][0], w * 16, 64, 32);

  v8f O[4];
  float mrow[8], lrow[8];
#pragma unroll
  for (int j = 0; j < 4; ++j)
#pragma unroll
    for (int e = 0; e < 8; ++e) O[j][e] = 0.f;
#pragma unroll
  for (int r = 0; r < 8; ++r) { mrow[r] = -1e30f; lrow[r] = 0.f; }

  int ntiles = (i0 >> 6) + 9;
  if (ntiles > (kKLEN >> 6)) ntiles = kKLEN >> 6;

  for (int tile = 0; tile < ntiles; ++tile) {
    const int j0 = tile * 64;
    __syncthreads();  // protect LDS from previous iteration readers
    {  // stage K (async b128) and V (transposed, scalar)
      const int jj = t >> 1, half = t & 1;
      const size_t kbase =
          ((size_t)(j0 + jj) * kB + b) * 3072 + 1024 + h * 64 + half * 32;
#pragma unroll
      for (int q4 = 0; q4 < 4; ++q4)
        cp_async_b128(&sK[jj][half * 32 + q4 * 8], qkv + kbase + q4 * 8);
      const size_t vbase =
          ((size_t)(j0 + jj) * kB + b) * 3072 + 2048 + h * 64 + half * 32;
#pragma unroll
      for (int dd = 0; dd < 32; ++dd)
        sVt[half * 32 + dd][jj] = qkv[vbase + dd];
    }
    {  // stage the 128-row r band: m = j0 + (Q-1) - i0 - 63 + t (async b128)
      int m = j0 + (kQ - 1) - i0 - 63 + t;
      if (m < 0) m = 0;
      if (m > kKLEN - 1) m = kKLEN - 1;  // out-of-range rows are masked anyway
      const __bf16* g = rproj + (size_t)m * 1024 + h * 64;
#pragma unroll
      for (int q4 = 0; q4 < 8; ++q4)
        cp_async_b128(&sR[t][q4 * 8], g + q4 * 8);
    }
    async_wait();
    __syncthreads();

    // ---- S = (q+u) @ K^T ----
    v8f S[4];
#pragma unroll
    for (int j = 0; j < 4; ++j) {
#pragma unroll
      for (int e = 0; e < 8; ++e) S[j][e] = 0.f;
      S[j] = wmma_bf16(quf[0], frag_b(&sK[0][0], j * 16, 64, 0), S[j]);
      S[j] = wmma_bf16(quf[1], frag_b(&sK[0][0], j * 16, 64, 32), S[j]);
    }
    // ---- T = (q+v) @ band^T  (16x128), bounce through LDS ----
#pragma unroll
    for (int ts = 0; ts < 8; ++ts) {
      v8f T;
#pragma unroll
      for (int e = 0; e < 8; ++e) T[e] = 0.f;
      T = wmma_bf16(qvf[0], frag_b(&sR[0][0], ts * 16, 64, 0), T);
      T = wmma_bf16(qvf[1], frag_b(&sR[0][0], ts * 16, 64, 32), T);
#pragma unroll
      for (int r = 0; r < 8; ++r)
        sT[w][r + 8 * hi][ts * 16 + lr] = T[r];
    }
    __syncthreads();

    // ---- fold shifted BD, scale, mask ----
#pragma unroll
    for (int j = 0; j < 4; ++j)
#pragma unroll
      for (int r = 0; r < 8; ++r) {
        const int ii = r + 8 * hi;
        const int jj = j * 16 + lr;
        const int tt = 63 - w * 16 + jj - ii;  // in [0,127]
        float s = (S[j][r] + sT[w][ii][tt]) * kSCALE;
        const int gi = i0 + w * 16 + ii;
        const int gj = j0 + jj;
        if (gj > gi + kMEM) s = -1e30f;
        S[j][r] = s;
      }

    // ---- online softmax (per-row over 64 keys) ----
    float mnew[8], sc[8], tsum[8];
#pragma unroll
    for (int r = 0; r < 8; ++r) {
      float tm = fmaxf(fmaxf(S[0][r], S[1][r]), fmaxf(S[2][r], S[3][r]));
      tm = redmax16(tm);
      mnew[r] = fmaxf(mrow[r], tm);
      sc[r] = __expf(mrow[r] - mnew[r]);
      mrow[r] = mnew[r];
      tsum[r] = 0.f;
    }
#pragma unroll
    for (int j = 0; j < 4; ++j)
#pragma unroll
      for (int r = 0; r < 8; ++r) {
        float p = __expf(S[j][r] - mnew[r]);
        S[j][r] = p;
        tsum[r] += p;
      }
#pragma unroll
    for (int r = 0; r < 8; ++r) {
      tsum[r] = redsum16(tsum[r]);
      lrow[r] = lrow[r] * sc[r] + tsum[r];
    }
#pragma unroll
    for (int j = 0; j < 4; ++j)
#pragma unroll
      for (int r = 0; r < 8; ++r) O[j][r] *= sc[r];

    // ---- P -> bf16 A-frag via LDS bounce; O += P @ V ----
#pragma unroll
    for (int j = 0; j < 4; ++j)
#pragma unroll
      for (int r = 0; r < 8; ++r)
        sP[w][r + 8 * hi][j * 16 + lr] = (__bf16)S[j][r];
    __syncthreads();
    v16bf pf0 = frag_a(&sP[w][0][0], 0, 64, 0);
    v16bf pf1 = frag_a(&sP[w][0][0], 0, 64, 32);
#pragma unroll
    for (int j = 0; j < 4; ++j) {
      O[j] = wmma_bf16(pf0, frag_b(&sVt[0][0], j * 16, 64, 0), O[j]);
      O[j] = wmma_bf16(pf1, frag_b(&sVt[0][0], j * 16, 64, 32), O[j]);
    }
  }

  // ---- normalize and store ctx (bf16) ----
#pragma unroll
  for (int j = 0; j < 4; ++j)
#pragma unroll
    for (int r = 0; r < 8; ++r) {
      const int i = i0 + w * 16 + r + 8 * hi;
      const int d = j * 16 + lr;
      const float o = O[j][r] / lrow[r];
      ctx[((size_t)i * kB + b) * 1024 + h * 64 + d] = (__bf16)o;
    }
}

// ---------------------------------------------------------------------------
// Residual add + LayerNorm over DM=1024. One block (256 thr) per row.
// ---------------------------------------------------------------------------
template <int WRITE_BF>
__global__ __launch_bounds__(256) void ln_k(
    const float* __restrict__ a, const float* __restrict__ bsrc,
    const float* __restrict__ g, const float* __restrict__ beta,
    float* __restrict__ xout, __bf16* __restrict__ xbf) {
  __shared__ float red[256];
  const int row = blockIdx.x;
  const int t = threadIdx.x;
  float vloc[4];
  float s = 0.f;
#pragma unroll
  for (int e = 0; e < 4; ++e) {
    const int c = t + e * 256;
    const float x = a[(size_t)row * 1024 + c] + bsrc[(size_t)row * 1024 + c];
    vloc[e] = x;
    s += x;
  }
  red[t] = s;
  __syncthreads();
  for (int o = 128; o > 0; o >>= 1) {
    if (t < o) red[t] += red[t + o];
    __syncthreads();
  }
  const float mean = red[0] * (1.f / 1024.f);
  __syncthreads();
  float s2 = 0.f;
#pragma unroll
  for (int e = 0; e < 4; ++e) {
    const float d = vloc[e] - mean;
    s2 += d * d;
  }
  red[t] = s2;
  __syncthreads();
  for (int o = 128; o > 0; o >>= 1) {
    if (t < o) red[t] += red[t + o];
    __syncthreads();
  }
  const float inv = rsqrtf(red[0] * (1.f / 1024.f) + 1e-5f);
#pragma unroll
  for (int e = 0; e < 4; ++e) {
    const int c = t + e * 256;
    const float y = (vloc[e] - mean) * inv * g[c] + beta[c];
    xout[(size_t)row * 1024 + c] = y;
    if (WRITE_BF) xbf[(size_t)row * 1024 + c] = (__bf16)y;
  }
}

// ---------------------------------------------------------------------------
// Host-side orchestration
// ---------------------------------------------------------------------------
extern "C" void kernel_launch(void* const* d_in, const int* in_sizes, int n_in,
                              void* d_out, int out_size, void* d_ws,
                              size_t ws_size, hipStream_t stream) {
  (void)in_sizes; (void)n_in; (void)out_size; (void)ws_size;
  const float* inputs = (const float*)d_in[0];
  const float* r      = (const float*)d_in[1];
  const float* u      = (const float*)d_in[2];
  const float* v      = (const float*)d_in[3];
  const float* mem    = (const float*)d_in[4];
  /* d_in[5] attn_mask: recomputed analytically (j > i + MEM) */
  const float* W_qkv  = (const float*)d_in[6];
  const float* W_r    = (const float*)d_in[7];
  const float* W_o    = (const float*)d_in[8];
  const float* ln1_g  = (const float*)d_in[9];
  const float* ln1_b  = (const float*)d_in[10];
  const float* ln2_g  = (const float*)d_in[11];
  const float* ln2_b  = (const float*)d_in[12];
  const float* W1     = (const float*)d_in[13];
  const float* b1     = (const float*)d_in[14];
  const float* W2     = (const float*)d_in[15];
  const float* b2     = (const float*)d_in[16];
  float* out = (float*)d_out;

  char* ws = (char*)d_ws;
  size_t off = 0;
  auto alloc = [&](size_t bytes) -> void* {
    void* p = ws + off;
    off += (bytes + 255) & ~(size_t)255;
    return p;
  };
  const size_t nCat  = (size_t)kKLEN * kB * kDM;     // 6.29M
  const size_t nR    = (size_t)kKLEN * kDM;          // 1.57M
  const size_t nQKV  = (size_t)kKLEN * kB * 3 * kDM; // 18.9M
  const size_t nRows = (size_t)kQ * kB;              // 4096
  __bf16* catB   = (__bf16*)alloc(nCat * 2);
  __bf16* rB     = (__bf16*)alloc(nR * 2);
  __bf16* WqkvT  = (__bf16*)alloc((size_t)kDM * 3 * kDM * 2);  // [3072][1024]
  __bf16* WrT    = (__bf16*)alloc((size_t)kDM * kDM * 2);      // [1024][1024]
  __bf16* WoT    = (__bf16*)alloc((size_t)kDM * kDM * 2);      // [1024][1024]
  __bf16* W1T    = (__bf16*)alloc((size_t)kDM * kMLP * 2);     // [4096][1024]
  __bf16* W2T    = (__bf16*)alloc((size_t)kMLP * kDM * 2);     // [1024][4096]
  __bf16* qkvB   = (__bf16*)alloc(nQKV * 2);
  __bf16* rprojB = (__bf16*)alloc(nR * 2);
  __bf16* ctxB   = (__bf16*)alloc(nRows * kDM * 2);
  float*  yF     = (float*)alloc(nRows * kDM * 4);   // reused for y2
  float*  xF     = (float*)alloc(nRows * kDM * 4);
  __bf16* xB     = (__bf16*)alloc(nRows * kDM * 2);
  __bf16* hB     = (__bf16*)alloc(nRows * (size_t)kMLP * 2);

  auto cvt = [&](const float* src, __bf16* dst, size_t n) {
    cvt_bf16_k<<<(unsigned)((n + 255) / 256), 256, 0, stream>>>(src, dst, (int)n);
  };
  auto cvtT = [&](const float* src, __bf16* dst, int Krows, int N) {
    transpose_cvt_k<<<dim3(N / 32, Krows / 32), 256, 0, stream>>>(src, dst,
                                                                  Krows, N);
  };
  // cat = [mem ; inputs] in bf16 (contiguous row-major [KLEN*B, DM])
  cvt(mem,    catB,                           (size_t)kMEM * kB * kDM);
  cvt(inputs, catB + (size_t)kMEM * kB * kDM, (size_t)kQ * kB * kDM);
  cvt(r,      rB, nR);
  // weights: convert + transpose to [N][K] so GEMM staging is contiguous
  cvtT(W_qkv, WqkvT, kDM, 3 * kDM);
  cvtT(W_r,   WrT,   kDM, kDM);
  cvtT(W_o,   WoT,   kDM, kDM);
  cvtT(W1,    W1T,   kDM, kMLP);
  cvtT(W2,    W2T,   kMLP, kDM);

  // qkv = cat @ W_qkv : [6144,1024]x[1024,3072] -> bf16
  gemm_bf16_k<1, 0, 0><<<dim3(3 * kDM / 128, kKLEN * kB / 128), 256, 0, stream>>>(
      catB, WqkvT, nullptr, nullptr, qkvB, kKLEN * kB, 3 * kDM, kDM);
  // rproj = r @ W_r : [1536,1024]x[1024,1024] -> bf16
  gemm_bf16_k<1, 0, 0><<<dim3(kDM / 128, kKLEN / 128), 256, 0, stream>>>(
      rB, WrT, nullptr, nullptr, rprojB, kKLEN, kDM, kDM);

  // attention: grid (Q/64, B*H)
  attn_k<<<dim3(kQ / 64, kB * kH), 128, 0, stream>>>(qkvB, rprojB, u, v, ctxB);

  // y = ctx @ W_o : [4096,1024]x[1024,1024] -> f32
  gemm_bf16_k<0, 0, 0><<<dim3(kDM / 128, (int)nRows / 128), 256, 0, stream>>>(
      ctxB, WoT, nullptr, yF, nullptr, (int)nRows, kDM, kDM);

  // x = LN(inputs + y) -> f32 + bf16
  ln_k<1><<<(unsigned)nRows, 256, 0, stream>>>(inputs, yF, ln1_g, ln1_b, xF, xB);

  // h = relu(x @ W1 + b1) : [4096,1024]x[1024,4096] -> bf16
  gemm_bf16_k<1, 1, 1><<<dim3(kMLP / 128, (int)nRows / 128), 256, 0, stream>>>(
      xB, W1T, b1, nullptr, hB, (int)nRows, kMLP, kDM);
  // y2 = h @ W2 + b2 : [4096,4096]x[4096,1024] -> f32 (reuse yF)
  gemm_bf16_k<0, 1, 0><<<dim3(kDM / 128, (int)nRows / 128), 256, 0, stream>>>(
      hB, W2T, b2, yF, nullptr, (int)nRows, kDM, kMLP);

  // out = LN(x + y2)
  ln_k<0><<<(unsigned)nRows, 256, 0, stream>>>(xF, yF, ln2_g, ln2_b, out, nullptr);
}
